// CrossAttention_30562987278517
// MI455X (gfx1250) — matrix operand order repaired and checked by reading, not compiled, not run on hardware
//
#include <hip/hip_runtime.h>
#include <hip/hip_bf16.h>

// ---------------------------------------------------------------------------
// Cross linear-attention, B=16, C=KC=VC=128, N=W*H=16384.
// All large matmuls via V_WMMA_F32_16X16X32_F16 (f16 in, f32 accum).
// Fragments fetched from LDS with ds_load_b128; tiles staged with
// GLOBAL_LOAD_ASYNC_TO_LDS_B128 (ASYNCcnt-tracked) when available.
// ---------------------------------------------------------------------------

typedef __attribute__((ext_vector_type(16))) _Float16 v16h;
typedef __attribute__((ext_vector_type(8)))  _Float16 v8h;
typedef __attribute__((ext_vector_type(8)))  float    v8f;

#define TILE_M 128
#define TILE_N 64
#define TILE_K 32
#define GEMM_THREADS 128   // 4 waves of 32; each wave computes 32(M) x 64(N)

#if defined(__has_builtin)
#  if __has_builtin(__builtin_amdgcn_global_load_async_to_lds_b128)
#    define USE_ASYNC_LDS 1
#  endif
#endif
#ifndef USE_ASYNC_LDS
#  define USE_ASYNC_LDS 0
#endif

#if USE_ASYNC_LDS
typedef int v4i __attribute__((vector_size(16)));
typedef __attribute__((address_space(1))) v4i* g4p;   // global b128 chunk
typedef __attribute__((address_space(3))) v4i* l4p;   // LDS b128 chunk

__device__ __forceinline__ void async_copy_b128(const _Float16* g, _Float16* l) {
    __builtin_amdgcn_global_load_async_to_lds_b128((g4p)g, (l4p)l, 0, 0);
}
__device__ __forceinline__ void wait_async_all() {
#  if __has_builtin(__builtin_amdgcn_s_wait_asynccnt)
    __builtin_amdgcn_s_wait_asynccnt(0);
#  else
    asm volatile("s_wait_asynccnt 0x0" ::: "memory");
#  endif
}
#endif

// Generic tiled WMMA GEMM:  Y[m,n] = sum_k A[m,k] * B(k,n) + bias[m]
//   A:  f16 row-major, leading dim lda (halves), per-batch stride aBatch
//   B:  if !transB: B(k,n) = Bm[k*ldb + n]; if transB: B(k,n) = Bm[n*ldb + k]
//   Y:  if !ytrans: Yb[m*ldy + n]; if ytrans: Yb[n*ldy + m]
// Requirements: M % 128 == 0, Kdim % 32 == 0, N multiples of 64; all base
// pointers / leading dims 16-byte aligned (true for all call sites).
// Uniform control flow -> EXEC all ones around every WMMA (ISA requirement).
template <typename OutT>
__global__ __launch_bounds__(GEMM_THREADS)
void wmma_gemm(const _Float16* __restrict__ A, long long aBatch, int lda,
               const _Float16* __restrict__ Bm, long long bBatch, int ldb, int transB,
               const float* __restrict__ bias, long long biasBatch,
               OutT* __restrict__ Y, long long yBatch, int ldy, int ytrans,
               int M, int Kdim)
{
    // Row pitch = 40 halves = 80 bytes = 5 * 16B: every 8-half chunk stays
    // 16-byte aligned and consecutive rows land on different bank groups.
    __shared__ __align__(16) _Float16 sA [TILE_M][TILE_K + 8];
    __shared__ __align__(16) _Float16 sBt[TILE_N][TILE_K + 8];   // transposed: [n][k]

    const int b  = blockIdx.z;
    const int m0 = blockIdx.y * TILE_M;
    const int n0 = blockIdx.x * TILE_N;

    const _Float16* Ab    = A  + (long long)b * aBatch;
    const _Float16* Bb    = Bm + (long long)b * bBatch;
    const float*    biasb = bias ? (bias + (long long)b * biasBatch) : nullptr;
    OutT*           Yb    = Y  + (long long)b * yBatch;

    const int tid   = threadIdx.x;
    const int wave  = tid >> 5;     // 0..3 -> 32-row M sub-tile
    const int lane  = tid & 31;
    const int lhalf = lane >> 4;    // 0 or 1 (lane group)
    const int l16   = lane & 15;

    v8f acc[2][4];
#pragma unroll
    for (int t = 0; t < 2; ++t)
#pragma unroll
        for (int nt = 0; nt < 4; ++nt)
#pragma unroll
            for (int j = 0; j < 8; ++j) acc[t][nt][j] = 0.0f;

    for (int k0 = 0; k0 < Kdim; k0 += TILE_K) {
        // ---- stage A tile: TILE_M x TILE_K, 4 x b128 chunks per thread ----
#pragma unroll
        for (int j = 0; j < 4; ++j) {
            int chunk = tid + GEMM_THREADS * j;          // 512 chunks total
            int r = chunk >> 2, c8 = (chunk & 3) * 8;    // 4 chunks per row
            const _Float16* gsrc = &Ab[(long long)(m0 + r) * lda + (k0 + c8)];
#if USE_ASYNC_LDS
            async_copy_b128(gsrc, &sA[r][c8]);
#else
            *(v8h*)&sA[r][c8] = *(const v8h*)gsrc;
#endif
        }
        // ---- stage B tile transposed into sBt[n][k] ----
        if (transB) {
            // Bm[n*ldb + k]: contiguous along k -> straight b128 copies
#pragma unroll
            for (int j = 0; j < 2; ++j) {
                int chunk = tid + GEMM_THREADS * j;      // 256 chunks total
                int n = chunk >> 2, k8 = (chunk & 3) * 8;
                const _Float16* gsrc = &Bb[(long long)(n0 + n) * ldb + (k0 + k8)];
#if USE_ASYNC_LDS
                async_copy_b128(gsrc, &sBt[n][k8]);
#else
                *(v8h*)&sBt[n][k8] = *(const v8h*)gsrc;
#endif
            }
        } else {
            // Bm[k*ldb + n]: contiguous along n -> b128 load, scatter 8 b16
#pragma unroll
            for (int j = 0; j < 2; ++j) {
                int chunk = tid + GEMM_THREADS * j;
                int k = chunk >> 3, n8 = (chunk & 7) * 8;
                v8h v = *(const v8h*)&Bb[(long long)(k0 + k) * ldb + (n0 + n8)];
#pragma unroll
                for (int e = 0; e < 8; ++e) sBt[n8 + e][k] = v[e];
            }
        }
        // prefetch next K-tile of B (emits global_prefetch_b8)
        if (k0 + TILE_K < Kdim) {
            long long pidx = transB ? ((long long)(n0 + (tid & 63)) * ldb + (k0 + TILE_K))
                                    : ((long long)(k0 + TILE_K + (tid >> 2)) * ldb + n0);
            __builtin_prefetch((const void*)&Bb[pidx], 0, 0);
        }
#if USE_ASYNC_LDS
        wait_async_all();           // ASYNCcnt -> 0 before the barrier
#endif
        __syncthreads();

        // ---- A fragments (16x32 layout): halves 0-7 = K[8*lhalf..),
        //      halves 8-15 = K[16+8*lhalf..) -> two ds_load_b128 per frag ----
        v16h af[2];
#pragma unroll
        for (int t = 0; t < 2; ++t) {
            const int arow = wave * 32 + t * 16 + l16;
            v8h a0 = *(const v8h*)&sA[arow][lhalf * 8];
            v8h a1 = *(const v8h*)&sA[arow][lhalf * 8 + 16];
            af[t] = __builtin_shufflevector(a0, a1, 0, 1, 2, 3, 4, 5, 6, 7,
                                            8, 9, 10, 11, 12, 13, 14, 15);
        }

#pragma unroll
        for (int nt = 0; nt < 4; ++nt) {
            // ---- B fragment (32x16): halves 0-7 = K[8*lhalf..), 8-15 = K[16+8*lhalf..) at col ----
            const int bcol = nt * 16 + l16;
            v8h b0 = *(const v8h*)&sBt[bcol][lhalf * 8];
            v8h b1 = *(const v8h*)&sBt[bcol][lhalf * 8 + 16];
            v16h bf = __builtin_shufflevector(b0, b1, 0, 1, 2, 3, 4, 5, 6, 7,
                                              8, 9, 10, 11, 12, 13, 14, 15);
#pragma unroll
            for (int t = 0; t < 2; ++t)
                acc[t][nt] = __builtin_amdgcn_wmma_f32_16x16x32_f16(
                    false, af[t], false, bf, (short)0, acc[t][nt], false, false);
        }
        __syncthreads();
    }

    // ---- epilogue: C/D layout: lane L, float j -> m = j + 8*lhalf, n = l16 ----
#pragma unroll
    for (int t = 0; t < 2; ++t) {
#pragma unroll
        for (int nt = 0; nt < 4; ++nt) {
            const int gn = n0 + nt * 16 + l16;
#pragma unroll
            for (int j = 0; j < 8; ++j) {
                const int gm = m0 + wave * 32 + t * 16 + j + (lhalf ? 8 : 0);
                float v = acc[t][nt][j] + (biasb ? biasb[gm] : 0.0f);
                if (ytrans) Yb[(long long)gn * ldy + gm] = (OutT)v;
                else        Yb[(long long)gm * ldy + gn] = (OutT)v;
            }
        }
    }
}

// f32 -> f16 conversion, 8 elements per thread (b128 traffic both sides).
__global__ void f32_to_f16_v8(const float* __restrict__ x, _Float16* __restrict__ y,
                              long long n8) {
    long long i = (long long)blockIdx.x * blockDim.x + threadIdx.x;
    if (i < n8) {
        v8f v = *(const v8f*)(x + i * 8);
        v8h h;
#pragma unroll
        for (int e = 0; e < 8; ++e) h[e] = (_Float16)v[e];
        *(v8h*)(y + i * 8) = h;
    }
}

// In-place L2 normalize over channel dim of [B, C, N] f16 (per column n).
__global__ void l2norm_inplace(_Float16* __restrict__ x, int Cc, long long Nn,
                               long long total) {
    long long col = (long long)blockIdx.x * blockDim.x + threadIdx.x;
    if (col >= total) return;
    long long bb = col / Nn, n = col % Nn;
    _Float16* p = x + bb * (long long)Cc * Nn + n;
    float s = 0.0f;
    for (int c = 0; c < Cc; ++c) { float v = (float)p[(long long)c * Nn]; s += v * v; }
    float r = rsqrtf(s);
    for (int c = 0; c < Cc; ++c)
        p[(long long)c * Nn] = (_Float16)((float)p[(long long)c * Nn] * r);
}

// Row sum: out[row] = sum_n x[row, n]; one block per row, v8h streaming.
__global__ void rowsum(const _Float16* __restrict__ x, float* __restrict__ out,
                       long long Nn) {
    long long row = blockIdx.x;
    const v8h* p8 = (const v8h*)(x + row * Nn);
    long long nv = Nn / 8;
    float s = 0.0f;
    for (long long i = threadIdx.x; i < nv; i += blockDim.x) {
        v8h v = p8[i];
#pragma unroll
        for (int e = 0; e < 8; ++e) s += (float)v[e];
    }
    __shared__ float red[256];
    red[threadIdx.x] = s;
    __syncthreads();
    for (int o = 128; o > 0; o >>= 1) {
        if (threadIdx.x < (unsigned)o) red[threadIdx.x] += red[threadIdx.x + o];
        __syncthreads();
    }
    if (threadIdx.x == 0) out[row] = red[0];
}

// den[n] = 1/(N + sum_c Qden[c,n]*(Ksum[c]+eps)); num[c,n] *= den[n] in place.
__global__ void den_scale(const _Float16* __restrict__ Qden,
                          const float* __restrict__ Ksum,
                          _Float16* __restrict__ num,
                          int Cc, long long Nn, float eps) {
    __shared__ float ks[128];
    const long long b = blockIdx.y;
    if ((int)threadIdx.x < Cc) ks[threadIdx.x] = Ksum[b * Cc + threadIdx.x] + eps;
    __syncthreads();
    long long n = (long long)blockIdx.x * blockDim.x + threadIdx.x;
    const _Float16* q = Qden + b * (long long)Cc * Nn + n;
    _Float16*       p = num  + b * (long long)Cc * Nn + n;
    float acc = (float)Nn;
    for (int c = 0; c < Cc; ++c) acc += (float)q[(long long)c * Nn] * ks[c];
    float d = 1.0f / acc;
    for (int c = 0; c < Cc; ++c)
        p[(long long)c * Nn] = (_Float16)((float)p[(long long)c * Nn] * d);
}

extern "C" void kernel_launch(void* const* d_in, const int* in_sizes, int n_in,
                              void* d_out, int out_size, void* d_ws, size_t ws_size,
                              hipStream_t stream) {
    (void)in_sizes; (void)n_in; (void)out_size; (void)ws_size;
    constexpr int       Bn = 16, Cc = 128;
    constexpr long long Nn = 128LL * 128LL;           // 16384
    constexpr long long CN = (long long)Cc * Nn;      // per-batch [C,N]
    constexpr long long T  = (long long)Bn * CN;      // full tensor elems
    constexpr float     EPS = 1e-6f;

    // inputs (setup_inputs order)
    const float* tensor1 = (const float*)d_in[0];
    const float* tensor2 = (const float*)d_in[1];
    const float* wq1 = (const float*)d_in[2];  const float* bq1 = (const float*)d_in[3];
    const float* wk1 = (const float*)d_in[4];  const float* bk1 = (const float*)d_in[5];
    const float* wv1 = (const float*)d_in[6];  const float* bv1 = (const float*)d_in[7];
    const float* wq2 = (const float*)d_in[8];  const float* bq2 = (const float*)d_in[9];
    const float* wk2 = (const float*)d_in[10]; const float* bk2 = (const float*)d_in[11];
    const float* wv2 = (const float*)d_in[12]; const float* bv2 = (const float*)d_in[13];
    const float* wr1 = (const float*)d_in[14]; const float* br1 = (const float*)d_in[15];
    const float* wr2 = (const float*)d_in[16]; const float* br2 = (const float*)d_in[17];
    const float* wcat = (const float*)d_in[18]; const float* bcat = (const float*)d_in[19];
    float* out = (float*)d_out;

    // workspace layout (f16 elements)
    _Float16* w    = (_Float16*)d_ws;
    _Float16* Xh1  = w;            // reused later as num1 / wv1
    _Float16* Xh2  = w + T;        // reused later as num2 / wv2
    _Float16* Qn1  = w + 2 * T;
    _Float16* Kh1  = w + 3 * T;
    _Float16* Vh1  = w + 4 * T;
    _Float16* Qn2  = w + 5 * T;
    _Float16* Kh2  = w + 6 * T;
    _Float16* Vh2  = w + 7 * T;
    _Float16* attcat = w + 8 * T;  // [B, 2C, N] -> 2*T elems
    _Float16* matT1  = w + 10 * T;            // [B,128,128]
    _Float16* matT2  = matT1 + (long long)Bn * 128 * 128;
    _Float16* wq1h = matT2 + (long long)Bn * 128 * 128;
    _Float16* wk1h = wq1h + 16384;
    _Float16* wv1h = wk1h + 16384;
    _Float16* wq2h = wv1h + 16384;
    _Float16* wk2h = wq2h + 16384;
    _Float16* wv2h = wk2h + 16384;
    _Float16* wr1h = wv2h + 16384;
    _Float16* wr2h = wr1h + 16384;
    _Float16* wcath = wr2h + 16384;           // 128x256
    float* fs    = (float*)(wcath + 32768);
    float* Ksum1 = fs;
    float* Vsum1 = fs + Bn * Cc;
    float* Ksum2 = fs + 2 * Bn * Cc;
    float* Vsum2 = fs + 3 * Bn * Cc;

    // ---- 1. convert inputs & weights to f16 (8 elems/thread) ----
    {
        int thr = 256;
        long long t8 = T / 8;
        f32_to_f16_v8<<<dim3((unsigned)((t8 + thr - 1) / thr)), thr, 0, stream>>>(tensor1, Xh1, t8);
        f32_to_f16_v8<<<dim3((unsigned)((t8 + thr - 1) / thr)), thr, 0, stream>>>(tensor2, Xh2, t8);
        f32_to_f16_v8<<<8, thr, 0, stream>>>(wq1, wq1h, 2048);
        f32_to_f16_v8<<<8, thr, 0, stream>>>(wk1, wk1h, 2048);
        f32_to_f16_v8<<<8, thr, 0, stream>>>(wv1, wv1h, 2048);
        f32_to_f16_v8<<<8, thr, 0, stream>>>(wq2, wq2h, 2048);
        f32_to_f16_v8<<<8, thr, 0, stream>>>(wk2, wk2h, 2048);
        f32_to_f16_v8<<<8, thr, 0, stream>>>(wv2, wv2h, 2048);
        f32_to_f16_v8<<<8, thr, 0, stream>>>(wr1, wr1h, 2048);
        f32_to_f16_v8<<<8, thr, 0, stream>>>(wr2, wr2h, 2048);
        f32_to_f16_v8<<<16, thr, 0, stream>>>(wcat, wcath, 4096);
    }

    const dim3 gProj((unsigned)(Nn / TILE_N), Cc / TILE_M, Bn);  // 256 x 1 x 16
    // ---- 2. six projection GEMMs: W[128,128] @ X[128,N] + b ----
    wmma_gemm<_Float16><<<gProj, GEMM_THREADS, 0, stream>>>(
        wq1h, 0, Cc, Xh1, CN, (int)Nn, 0, bq1, 0, Qn1, CN, (int)Nn, 0, Cc, Cc);
    wmma_gemm<_Float16><<<gProj, GEMM_THREADS, 0, stream>>>(
        wk1h, 0, Cc, Xh1, CN, (int)Nn, 0, bk1, 0, Kh1, CN, (int)Nn, 0, Cc, Cc);
    wmma_gemm<_Float16><<<gProj, GEMM_THREADS, 0, stream>>>(
        wv1h, 0, Cc, Xh1, CN, (int)Nn, 0, bv1, 0, Vh1, CN, (int)Nn, 0, Cc, Cc);
    wmma_gemm<_Float16><<<gProj, GEMM_THREADS, 0, stream>>>(
        wq2h, 0, Cc, Xh2, CN, (int)Nn, 0, bq2, 0, Qn2, CN, (int)Nn, 0, Cc, Cc);
    wmma_gemm<_Float16><<<gProj, GEMM_THREADS, 0, stream>>>(
        wk2h, 0, Cc, Xh2, CN, (int)Nn, 0, bk2, 0, Kh2, CN, (int)Nn, 0, Cc, Cc);
    wmma_gemm<_Float16><<<gProj, GEMM_THREADS, 0, stream>>>(
        wv2h, 0, Cc, Xh2, CN, (int)Nn, 0, bv2, 0, Vh2, CN, (int)Nn, 0, Cc, Cc);

    // ---- 3. L2-normalize Q and K over channels ----
    {
        long long cols = (long long)Bn * Nn;
        unsigned g = (unsigned)((cols + 255) / 256);
        l2norm_inplace<<<g, 256, 0, stream>>>(Qn1, Cc, Nn, cols);
        l2norm_inplace<<<g, 256, 0, stream>>>(Kh1, Cc, Nn, cols);
        l2norm_inplace<<<g, 256, 0, stream>>>(Qn2, Cc, Nn, cols);
        l2norm_inplace<<<g, 256, 0, stream>>>(Kh2, Cc, Nn, cols);
    }

    // ---- 4. row sums: Ksum[b,c], Vsum[b,c] ----
    rowsum<<<Bn * Cc, 256, 0, stream>>>(Kh1, Ksum1, Nn);
    rowsum<<<Bn * Cc, 256, 0, stream>>>(Vh1, Vsum1, Nn);
    rowsum<<<Bn * Cc, 256, 0, stream>>>(Kh2, Ksum2, Nn);
    rowsum<<<Bn * Cc, 256, 0, stream>>>(Vh2, Vsum2, Nn);

    // ---- 5. mat = K @ V^T  (K-dim = N), stored transposed: matT[c,m] ----
    const dim3 gMat(128 / TILE_N, 128 / TILE_M, Bn);  // 2 x 1 x 16
    wmma_gemm<_Float16><<<gMat, GEMM_THREADS, 0, stream>>>(
        Kh1, CN, (int)Nn, Vh1, CN, (int)Nn, 1, nullptr, 0,
        matT1, 128LL * 128, 128, 1, Cc, (int)Nn);
    wmma_gemm<_Float16><<<gMat, GEMM_THREADS, 0, stream>>>(
        Kh2, CN, (int)Nn, Vh2, CN, (int)Nn, 1, nullptr, 0,
        matT2, 128LL * 128, 128, 1, Cc, (int)Nn);

    // ---- 6. num[c,n] = Vsum[c] + matT @ Qn  (reuse Xh buffers) ----
    _Float16* num1 = Xh1;
    _Float16* num2 = Xh2;
    wmma_gemm<_Float16><<<gProj, GEMM_THREADS, 0, stream>>>(
        matT1, 128LL * 128, 128, Qn1, CN, (int)Nn, 0, Vsum1, Cc,
        num1, CN, (int)Nn, 0, Cc, Cc);
    wmma_gemm<_Float16><<<gProj, GEMM_THREADS, 0, stream>>>(
        matT2, 128LL * 128, 128, Qn2, CN, (int)Nn, 0, Vsum2, Cc,
        num2, CN, (int)Nn, 0, Cc, Cc);

    // ---- 7. den + scale (branch2 den uses Q1, branch1 den uses Q2) ----
    {
        dim3 g((unsigned)(Nn / 128), Bn);
        den_scale<<<g, 128, 0, stream>>>(Qn1, Ksum2, num2, Cc, Nn, EPS);
        den_scale<<<g, 128, 0, stream>>>(Qn2, Ksum1, num1, Cc, Nn, EPS);
    }

    // ---- 8. att1 = wr2 @ wv1 + br2 -> attcat rows [0,128)
    //          att2 = wr1 @ wv2 + br1 -> attcat rows [128,256) ----
    wmma_gemm<_Float16><<<gProj, GEMM_THREADS, 0, stream>>>(
        wr2h, 0, Cc, num1, CN, (int)Nn, 0, br2, 0,
        attcat, 2 * CN, (int)Nn, 0, Cc, Cc);
    wmma_gemm<_Float16><<<gProj, GEMM_THREADS, 0, stream>>>(
        wr1h, 0, Cc, num2, CN, (int)Nn, 0, br1, 0,
        attcat + Cc * Nn, 2 * CN, (int)Nn, 0, Cc, Cc);

    // ---- 9. out = wcat[128,256] @ attcat[256,N] + bcat (f32 output) ----
    wmma_gemm<float><<<gProj, GEMM_THREADS, 0, stream>>>(
        wcath, 0, 2 * Cc, attcat, 2 * CN, (int)Nn, 0, bcat, 0,
        out, CN, (int)Nn, 0, Cc, 2 * Cc);
}
